// Attention_1073741824686
// MI455X (gfx1250) — matrix operand order repaired and checked by reading, not compiled, not run on hardware
//
#include <hip/hip_runtime.h>
#include <hip/hip_bf16.h>

// ---------------------------------------------------------------------------
// Fused causal MHA for MI455X (gfx1250, wave32, WMMA).
//   B=2, S=2048, D=1024, H=16, HD=64
// Pipeline:
//   0) one-time f32->f16 weight conversion into workspace (bandwidth-trivial)
//   1) qkv GEMM (WMMA f16, LDS B-tiles filled by global_load_async_to_lds)
//   2) per-(bh, 16-query-tile) causal attention with LDS-resident logit strip
//   3) out GEMM (same async-staged WMMA skeleton)
// ---------------------------------------------------------------------------

#define B_  2
#define S_  2048
#define D_  1024
#define H_  16
#define HD_ 64
#define N_QKV (3 * D_)

typedef __attribute__((ext_vector_type(16))) _Float16 v16h;
typedef __attribute__((ext_vector_type(8)))  float    v8f;
typedef __attribute__((ext_vector_type(4)))  int      v4i;

#if __has_builtin(__builtin_amdgcn_global_load_async_to_lds_b128)
#define USE_ASYNC_LDS 1
typedef __attribute__((address_space(1))) v4i GV4;   // global-AS int4
typedef __attribute__((address_space(3))) v4i LV4;   // LDS-AS int4
#else
#define USE_ASYNC_LDS 0
#endif

__device__ inline void wait_async0() {
#if __has_builtin(__builtin_amdgcn_s_wait_asynccnt)
  __builtin_amdgcn_s_wait_asynccnt(0);
#elif USE_ASYNC_LDS
  asm volatile("s_wait_asynccnt 0x0" ::: "memory");
#endif
}

// A-fragment (16x32, f16): lane L -> row m = L%16; lane-half selects K groups.
// Per ISA 7.12.2: lanes 0-15 hold K {0..7,16..23}, lanes 16-31 hold K {8..15,24..31}.
__device__ inline v16h load_a_f32(const float* __restrict__ row, int k0, int half) {
  v16h a;
#pragma unroll
  for (int j = 0; j < 8; ++j) a[j] = (_Float16)row[k0 + 8 * half + j];
#pragma unroll
  for (int j = 0; j < 8; ++j) a[8 + j] = (_Float16)row[k0 + 16 + 8 * half + j];
  return a;
}

__device__ inline v16h load_a_f16(const _Float16* __restrict__ row, int k0, int half) {
  v16h a;
#pragma unroll
  for (int j = 0; j < 8; ++j) a[j] = row[k0 + 8 * half + j];
#pragma unroll
  for (int j = 0; j < 8; ++j) a[8 + j] = row[k0 + 16 + 8 * half + j];
  return a;
}

// B-fragment (32x16, f16): lane L -> column N = L%16; vector elem j -> K index.
__device__ inline v16h load_b_f16(const _Float16* __restrict__ row16) {
  v16h b;
#pragma unroll
  for (int j = 0; j < 16; ++j) b[j] = row16[j];
  return b;
}

// Stage one 32(K) x 64(N) f16 B tile into LDS, fragment-major:
// dst[(c*32 + k)*16 + j] = W16[k0+k][n0 + c*16 + j].  128 threads: t = k + 32*c.
// Async path: two b128 async copies per thread, landing directly in the
// swizzled layout (per-lane LDS destinations); tracked by ASYNCcnt.
__device__ inline void stage_b_f16(const _Float16* __restrict__ W16, int ldn,
                                   int k0, int n0, _Float16* __restrict__ dst,
                                   int tid) {
  const int k = tid & 31;
  const int c = tid >> 5;
  const _Float16* src = W16 + (size_t)(k0 + k) * ldn + n0 + c * 16;
  _Float16* d = dst + (c * 32 + k) * 16;
#if USE_ASYNC_LDS
  __builtin_amdgcn_global_load_async_to_lds_b128(
      (GV4*)src, (LV4*)d, 0, 0);
  __builtin_amdgcn_global_load_async_to_lds_b128(
      (GV4*)(src + 8), (LV4*)(d + 8), 0, 0);
#else
#pragma unroll
  for (int j = 0; j < 16; ++j) d[j] = src[j];
#endif
}

// Read one B fragment: lane -> column N = lane%16; elem s -> K in table order.
__device__ inline v16h read_b_stage(const _Float16* __restrict__ buf, int c,
                                    int lane) {
  v16h b;
#pragma unroll
  for (int s = 0; s < 16; ++s) {
    const int k = s + ((s >= 8) ? 8 : 0) + 8 * (lane >> 4);
    b[s] = buf[(c * 32 + k) * 16 + (lane & 15)];
  }
  return b;
}

// ---------------------------------------------------------------------------
// Kernel 0: f32 -> f16 elementwise convert (weights into workspace).
// ---------------------------------------------------------------------------
__global__ __launch_bounds__(256) void cvt_kernel(const float* __restrict__ src,
                                                  _Float16* __restrict__ dst,
                                                  int n) {
  int i = blockIdx.x * blockDim.x + threadIdx.x;
  const int stride = gridDim.x * blockDim.x;
  for (; i < n; i += stride) dst[i] = (_Float16)src[i];
}

// ---------------------------------------------------------------------------
// Kernel 1: qkv = x @ W_qkv + b_qkv, scattered into Q [bh][s][hd] (f16),
// K^T [bh][hd][s] (f16, transposed for contiguous attention B-frags),
// V [bh][s][hd] (f16).
// Block: 128 threads = 4 waves stacked on M; wave tile 16x64; block 64x64.
// ---------------------------------------------------------------------------
__global__ __launch_bounds__(128) void qkv_kernel(
    const float* __restrict__ x, const _Float16* __restrict__ W16,
    const float* __restrict__ bqkv,
    _Float16* __restrict__ Qws, _Float16* __restrict__ Ktws,
    _Float16* __restrict__ Vws) {
  __shared__ __align__(16) _Float16 bstage[2][4 * 32 * 16];  // 2 x 4KB

  const int tid  = threadIdx.x;
  const int lane = tid & 31;
  const int wave = tid >> 5;
  const int half = lane >> 4;
  const int nl   = lane & 15;
  const int m0   = blockIdx.x * 64 + wave * 16;   // token row (flattened B*S)
  const int n0   = blockIdx.y * 64;               // qkv column

  v8f acc[4] = {};
  const float* arow = x + (size_t)(m0 + nl) * D_;

  stage_b_f16(W16, N_QKV, 0, n0, bstage[0], tid);
  wait_async0();
  __syncthreads();

  for (int k0 = 0; k0 < D_; k0 += 32) {
    const int cur = (k0 >> 5) & 1;
    if (k0 + 32 < D_) {
      stage_b_f16(W16, N_QKV, k0 + 32, n0, bstage[cur ^ 1], tid);
      __builtin_prefetch(arow + k0 + 64, 0, 1);
    }
    v16h a = load_a_f32(arow, k0, half);
#pragma unroll
    for (int c = 0; c < 4; ++c) {
      v16h bfrag = read_b_stage(bstage[cur], c, lane);
      acc[c] = __builtin_amdgcn_wmma_f32_16x16x32_f16(
          false, a, false, bfrag, (short)0, acc[c], false, false);
    }
    wait_async0();        // next tile's async copies landed in LDS
    __syncthreads();      // publish to all waves / protect re-staging
  }

  // Epilogue: bias add + scatter per head / Q-K-V part.
#pragma unroll
  for (int c = 0; c < 4; ++c) {
#pragma unroll
    for (int r = 0; r < 8; ++r) {
      const int ml = r + 8 * half;                // C-frag: VGPR r, lane half
      const int m  = m0 + ml;
      const int n  = n0 + c * 16 + nl;
      const float v = acc[c][r] + bqkv[n];
      const int b    = m / S_;
      const int s    = m % S_;
      const int head = n / (3 * HD_);
      const int part = (n / HD_) % 3;
      const int hd   = n % HD_;
      const int bh   = b * H_ + head;
      const _Float16 hv = (_Float16)v;
      if (part == 0)      Qws [((size_t)bh * S_  + s ) * HD_ + hd] = hv;
      else if (part == 1) Ktws[((size_t)bh * HD_ + hd) * S_  + s ] = hv;
      else                Vws [((size_t)bh * S_  + s ) * HD_ + hd] = hv;
    }
  }
}

// ---------------------------------------------------------------------------
// Kernel 2: causal attention for one (bh, 16-query tile) per workgroup.
// 8 waves. Full 16 x S f32 logit strip lives in LDS (CDNA5: 320KB/WGP).
// ---------------------------------------------------------------------------
#define SROW (S_ + 4)   // padded f32 row stride: lanes land on distinct banks

__global__ __launch_bounds__(256) void attn_kernel(
    const _Float16* __restrict__ Qws, const _Float16* __restrict__ Ktws,
    const _Float16* __restrict__ Vws, _Float16* __restrict__ attn_out) {
  __shared__ float scores[16 * SROW];      // 131,328 B
  __shared__ float partial[8 * 256];       //   8,192 B
  __shared__ float rmax[16], rinv[16];

  const int bh   = blockIdx.x;
  const int qt   = blockIdx.y;
  const int q0   = qt * 16;
  const int tid  = threadIdx.x;
  const int lane = tid & 31;
  const int wave = tid >> 5;
  const int half = lane >> 4;
  const int nl   = lane & 15;
  const float scale = 0.125f;              // 1/sqrt(HD)

  // ---- Phase 1: logits = scale * Q K^T with causal mask -> LDS ----
  const _Float16* Qb  = Qws  + ((size_t)bh * S_ + q0) * HD_;
  const _Float16* Ktb = Ktws + (size_t)bh * HD_ * S_;

  v16h aq[2];
#pragma unroll
  for (int c2 = 0; c2 < 2; ++c2)
    aq[c2] = load_a_f16(Qb + (size_t)nl * HD_, c2 * 32, half);

  for (int kt = wave; kt <= qt; kt += 8) {   // wave-uniform loop
    v8f acc = {};
#pragma unroll
    for (int c2 = 0; c2 < 2; ++c2) {
      v16h bf;
#pragma unroll
      for (int j = 0; j < 16; ++j) {
        const int kj = j + ((j >= 8) ? 8 : 0) + 8 * half;
        bf[j] = Ktb[(size_t)(c2 * 32 + kj) * S_ + kt * 16 + nl];
      }
      acc = __builtin_amdgcn_wmma_f32_16x16x32_f16(
          false, aq[c2], false, bf, (short)0, acc, false, false);
    }
#pragma unroll
    for (int r = 0; r < 8; ++r) {
      const int ml  = r + 8 * half;
      const int col = kt * 16 + nl;
      scores[ml * SROW + col] =
          (col <= q0 + ml) ? acc[r] * scale : -3.0e38f;
    }
  }
  __syncthreads();

  // ---- Phase 2: row max / sum-exp (16 threads per row) ----
  const int n_cols = (qt + 1) * 16;
  {
    const int row = tid >> 4;
    const int sub = tid & 15;
    float m = -3.0e38f;
    for (int col = sub; col < n_cols; col += 16)
      m = fmaxf(m, scores[row * SROW + col]);
#pragma unroll
    for (int s = 8; s >= 1; s >>= 1) m = fmaxf(m, __shfl_xor(m, s, 16));
    float sum = 0.0f;
    for (int col = sub; col < n_cols; col += 16)
      sum += __expf(scores[row * SROW + col] - m);
#pragma unroll
    for (int s = 8; s >= 1; s >>= 1) sum += __shfl_xor(sum, s, 16);
    if (sub == 0) { rmax[row] = m; rinv[row] = 1.0f / sum; }
  }
  __syncthreads();

  // ---- Phase 3: out = softmax(P) @ V, split-K across wave halves ----
  const int c  = wave & 3;                  // hd column tile (4 x 16 = 64)
  const int kh = wave >> 2;                 // key-range half
  const int chunks = (n_cols + 31) >> 5;
  const int mid = (chunks + 1) >> 1;
  const int cb = kh ? mid : 0;
  const int ce = kh ? chunks : mid;
  const float rm = rmax[nl];
  const float ri = rinv[nl];
  const _Float16* Vb = Vws + (size_t)bh * S_ * HD_;

  v8f acc = {};
  for (int ch = cb; ch < ce; ++ch) {        // wave-uniform loop
    const int kbase = ch * 32;
    v16h pa;
#pragma unroll
    for (int j = 0; j < 16; ++j) {
      const int kj  = j + ((j >= 8) ? 8 : 0) + 8 * half;
      const int col = kbase + kj;
      float p = 0.0f;
      if (col < n_cols)
        p = __expf(scores[nl * SROW + col] - rm) * ri;
      pa[j] = (_Float16)p;
    }
    v16h vb;
#pragma unroll
    for (int j = 0; j < 16; ++j) {
      const int kj = j + ((j >= 8) ? 8 : 0) + 8 * half;
      vb[j] = Vb[(size_t)(kbase + kj) * HD_ + c * 16 + nl];
    }
    acc = __builtin_amdgcn_wmma_f32_16x16x32_f16(
        false, pa, false, vb, (short)0, acc, false, false);
  }

#pragma unroll
  for (int r = 0; r < 8; ++r) partial[wave * 256 + r * 32 + lane] = acc[r];
  __syncthreads();

  if (wave < 4) {
    const int b = bh / H_, h = bh % H_;
#pragma unroll
    for (int r = 0; r < 8; ++r) {
      const float v = acc[r] + partial[(wave + 4) * 256 + r * 32 + lane];
      const int ml = r + 8 * half;
      const size_t idx =
          ((size_t)(b * S_ + q0 + ml)) * D_ + h * HD_ + c * 16 + nl;
      attn_out[idx] = (_Float16)v;
    }
  }
}

// ---------------------------------------------------------------------------
// Kernel 3: out = attn @ W_out + b_out  (A is f16 workspace, C is f32).
// ---------------------------------------------------------------------------
__global__ __launch_bounds__(128) void proj_kernel(
    const _Float16* __restrict__ A, const _Float16* __restrict__ W16,
    const float* __restrict__ bias, float* __restrict__ out) {
  __shared__ __align__(16) _Float16 bstage[2][4 * 32 * 16];

  const int tid  = threadIdx.x;
  const int lane = tid & 31;
  const int wave = tid >> 5;
  const int half = lane >> 4;
  const int nl   = lane & 15;
  const int m0   = blockIdx.x * 64 + wave * 16;
  const int n0   = blockIdx.y * 64;

  v8f acc[4] = {};
  const _Float16* arow = A + (size_t)(m0 + nl) * D_;

  stage_b_f16(W16, D_, 0, n0, bstage[0], tid);
  wait_async0();
  __syncthreads();

  for (int k0 = 0; k0 < D_; k0 += 32) {
    const int cur = (k0 >> 5) & 1;
    if (k0 + 32 < D_) {
      stage_b_f16(W16, D_, k0 + 32, n0, bstage[cur ^ 1], tid);
      __builtin_prefetch(arow + k0 + 64, 0, 1);
    }
    v16h a = load_a_f16(arow, k0, half);
#pragma unroll
    for (int c = 0; c < 4; ++c) {
      v16h bfrag = read_b_stage(bstage[cur], c, lane);
      acc[c] = __builtin_amdgcn_wmma_f32_16x16x32_f16(
          false, a, false, bfrag, (short)0, acc[c], false, false);
    }
    wait_async0();
    __syncthreads();
  }

#pragma unroll
  for (int c = 0; c < 4; ++c) {
#pragma unroll
    for (int r = 0; r < 8; ++r) {
      const int ml = r + 8 * half;
      const int n  = n0 + c * 16 + nl;
      out[(size_t)(m0 + ml) * D_ + n] = acc[c][r] + bias[n];
    }
  }
}

// ---------------------------------------------------------------------------
// Launch
// ---------------------------------------------------------------------------
extern "C" void kernel_launch(void* const* d_in, const int* in_sizes, int n_in,
                              void* d_out, int out_size, void* d_ws,
                              size_t ws_size, hipStream_t stream) {
  (void)in_sizes; (void)n_in; (void)out_size; (void)ws_size;
  const float* x     = (const float*)d_in[0];
  const float* Wqkv  = (const float*)d_in[1];
  const float* bqkv  = (const float*)d_in[2];
  const float* Wout  = (const float*)d_in[3];
  const float* bout  = (const float*)d_in[4];
  float* out = (float*)d_out;

  const size_t seg = (size_t)B_ * H_ * S_ * HD_;   // 4,194,304 elems (f16)
  _Float16* Qws   = (_Float16*)d_ws;
  _Float16* Ktws  = Qws  + seg;
  _Float16* Vws   = Ktws + seg;
  _Float16* attn  = Vws  + seg;                    // B*S*D f16
  _Float16* W16q  = attn + seg;                    // D * 3D f16
  _Float16* W16o  = W16q + (size_t)D_ * N_QKV;     // D * D  f16

  cvt_kernel<<<2048, 256, 0, stream>>>(Wqkv, W16q, D_ * N_QKV);
  cvt_kernel<<<2048, 256, 0, stream>>>(Wout, W16o, D_ * D_);
  qkv_kernel<<<dim3((B_ * S_) / 64, N_QKV / 64), 128, 0, stream>>>(
      x, W16q, bqkv, Qws, Ktws, Vws);
  attn_kernel<<<dim3(B_ * H_, S_ / 16), 256, 0, stream>>>(
      Qws, Ktws, Vws, attn);
  proj_kernel<<<dim3((B_ * S_) / 64, D_ / 64), 128, 0, stream>>>(
      attn, W16o, bout, out);
}
